// Region2Vec_3023656976611
// MI455X (gfx1250) — matrix operand order repaired and checked by reading, not compiled, not run on hardware
//
#include <hip/hip_runtime.h>
#include <hip/hip_bf16.h>

// ---------------------------------------------------------------------------
// GraphSAGE x2 (mean aggregate) + BN + ReLU + L2-norm, for MI455X (gfx1250).
// Dense GEMMs: v_wmma_f32_16x16x32_f16 with B tiles staged by the Tensor Data
// Mover (tensor_load_to_lds, double-buffered, padded LDS rows) when available.
// Sparse part: coalesced float4 gathers + global f32 atomics (HBM-bound).
// ---------------------------------------------------------------------------

typedef _Float16 v16h __attribute__((ext_vector_type(16)));
typedef _Float16 v8h  __attribute__((ext_vector_type(8)));
typedef float    v8f  __attribute__((ext_vector_type(8)));
typedef unsigned int u32x4 __attribute__((ext_vector_type(4)));
typedef int          i32x4 __attribute__((ext_vector_type(4)));
typedef int          i32x8 __attribute__((ext_vector_type(8)));

#define HIDN 256
#define AS_STRIDE 40   // halves; 80B rows -> 16B-aligned sub-chunks
#define BS_STRIDE 40   // halves; matches TDM pad: 16 dwords data + 4 dwords pad

#if defined(__has_builtin)
#  if __has_builtin(__builtin_amdgcn_tensor_load_to_lds) && \
      __has_builtin(__builtin_amdgcn_s_wait_tensorcnt)
#    define HAVE_TDM 1
#  endif
#endif
#ifndef HAVE_TDM
#  define HAVE_TDM 0
#endif

// ---------------- utility kernels ----------------

__global__ void zero_f32(float* __restrict__ p, long n) {
    long i = (long)blockIdx.x * blockDim.x + threadIdx.x;
    if (i < n) p[i] = 0.0f;
}

// Concat [W_l ; W_r] (f32, KxN row-major, N=256) -> f16 TRANSPOSED: out[n][Ktot].
__global__ void cvt_weights_t(const float* __restrict__ Wl, const float* __restrict__ Wr,
                              _Float16* __restrict__ out, int K1, int K2) {
    const int Ktot = K1 + K2;
    long n = (long)Ktot * HIDN;
    long i = (long)blockIdx.x * blockDim.x + threadIdx.x;
    if (i >= n) return;
    int k   = (int)(i / HIDN);
    int col = (int)(i % HIDN);
    float v = (k < K1) ? Wl[(long)k * HIDN + col] : Wr[(long)(k - K1) * HIDN + col];
    out[(long)col * Ktot + k] = (_Float16)v;
}

__global__ void edge_count(const long long* __restrict__ ei, float* __restrict__ cnt, int E) {
    int e = blockIdx.x * blockDim.x + threadIdx.x;
    if (e < E) atomicAdd(cnt + (int)ei[(long)E + e], 1.0f);
}

__global__ void make_rscale(const float* __restrict__ cnt, float* __restrict__ rs, int n) {
    int i = blockIdx.x * blockDim.x + threadIdx.x;
    if (i < n) rs[i] = 1.0f / fmaxf(cnt[i], 1.0f);
}

// agg[dst] += x[src]   (one thread per edge x 4 dims, float4 load + 4 atomics)
template <int DIM>
__global__ void scatter_add(const long long* __restrict__ ei, const float* __restrict__ x,
                            float* __restrict__ agg, int E) {
    const int VPE = DIM / 4;
    long t = (long)blockIdx.x * blockDim.x + threadIdx.x;
    long total = (long)E * VPE;
    if (t >= total) return;
    int e = (int)(t / VPE);
    int j = (int)(t % VPE) * 4;
    int src = (int)ei[e];
    int dst = (int)ei[(long)E + e];
    float4 v = *(const float4*)(x + (long)src * DIM + j);
    float* d = agg + (long)dst * DIM + j;
    atomicAdd(d + 0, v.x);
    atomicAdd(d + 1, v.y);
    atomicAdd(d + 2, v.z);
    atomicAdd(d + 3, v.w);
}

// ---------------- TDM issue: 2D weight tile (32 k x 256 n) -> padded LDS ----
#if HAVE_TDM
__device__ __forceinline__ void tdm_load_b_tile(const _Float16* gsrc, void* lds_dst, int Ktot) {
    // D# per CDNA5 ISA 8.3-8.5. Tensor = Wt[n][k], row stride Ktot elements.
    // Tile: tile_dim0 = 32 (k, 64B rows), tile_dim1 = 256 (n).
    // LDS pad: 4 dwords after every 16 dwords -> 80B row pitch == BS_STRIDE.
    unsigned long long ga = (unsigned long long)(uintptr_t)gsrc;
    unsigned lds = (unsigned)(uintptr_t)lds_dst;  // low 32 bits = LDS offset
    u32x4 g0;
    g0[0] = 1u;                                            // count=1 user descriptor
    g0[1] = lds;                                           // lds_addr
    g0[2] = (unsigned)(ga & 0xFFFFFFFFu);                  // global_addr[31:0]
    g0[3] = (unsigned)((ga >> 32) & 0x01FFFFFFu) | (2u << 30);  // addr[56:32] | type=2
    i32x8 g1;
    g1[0] = (1 << 16)      // data_size = 2 bytes
          | (1 << 20)      // pad_enable
          | (3 << 22)      // pad_interval: 16 dwords
          | (3 << 25);     // pad_amount:   4 dwords
    g1[1] = (Ktot & 0xFFFF) << 16;                         // tensor_dim0[15:0]
    g1[2] = ((Ktot >> 16) & 0xFFFF) | (HIDN << 16);        // tensor_dim0 hi | tensor_dim1 lo
    g1[3] = (32 << 16);                                    // tensor_dim1 hi | tile_dim0=32
    g1[4] = HIDN;                                          // tile_dim1=256, tile_dim2=0
    g1[5] = Ktot;                                          // tensor_dim0_stride[31:0]
    g1[6] = 0;                                             // stride hi | dim1_stride lo
    g1[7] = 0;
    i32x4 z4 = {0, 0, 0, 0};
    i32x8 z8 = {0, 0, 0, 0, 0, 0, 0, 0};
    // 6-arg form (amdgpu-toolchain / clang-23): (g0, g1, g2, g3, g4, cpol)
    __builtin_amdgcn_tensor_load_to_lds(g0, g1, z4, z4, z8, 0);
}
#endif

// ---------------- WMMA GEMM ----------------
// out[M x 256] = (A0 * rscale_row) @ W[0:K1] + A1 @ W[K1:K1+K2] + bias
// W is f16 TRANSPOSED: Wt[n][Ktot]. Block: 256 threads (8 waves) -> 64x256.
// wave w: m-tile = w>>1 (16 rows), n-half = w&1 (128 cols = 8 WMMA n-tiles).
__launch_bounds__(256)
__global__ void sage_gemm(const float* __restrict__ A0, int lda0,
                          const float* __restrict__ A1, int lda1,
                          const float* __restrict__ rscale,
                          const _Float16* __restrict__ W,
                          const float* __restrict__ bias,
                          float* __restrict__ out,
                          int K1, int K2, int nrows) {
    __shared__ _Float16 As[64 * AS_STRIDE];       // [m][k]  activations (f32->f16)
    __shared__ _Float16 Bs[2][256 * BS_STRIDE];   // [n][k]  weights, double buffered

    const int tid  = threadIdx.x;
    const int lane = tid & 31;
    const int wave = tid >> 5;
    const int mt   = wave >> 1;
    const int nh   = wave & 1;
    const int m    = lane & 15;
    const int hi   = lane >> 4;
    const int blockRow = blockIdx.x * 64;

    const v8f vzero = {0.f, 0.f, 0.f, 0.f, 0.f, 0.f, 0.f, 0.f};
    v8f acc[8];
#pragma unroll
    for (int i = 0; i < 8; i++) acc[i] = vzero;

    const int Ktot = K1 + K2;

#if HAVE_TDM
    if (wave == 0) tdm_load_b_tile(W, &Bs[0][0], Ktot);   // prime buffer 0
#endif

    for (int kb = 0; kb < Ktot; kb += 32) {
        const int    buf  = (kb >> 5) & 1;
        const bool   p0   = (kb < K1);
        const float* A    = p0 ? A0 : A1;
        const int    lda  = p0 ? lda0 : lda1;
        const int    kOff = p0 ? kb : (kb - K1);

        // --- stage A tile: 64 rows x 32 k (f32 -> f16, per-row scale on agg)
        {
            int r  = tid >> 2;
            int c0 = (tid & 3) * 8;
            int grow = blockRow + r;
            if (grow >= nrows) grow = nrows - 1;
            float s = p0 ? rscale[grow] : 1.0f;
            const float* ap = A + (long)grow * lda + kOff + c0;
            float4 va = *(const float4*)ap;
            float4 vb = *(const float4*)(ap + 4);
            _Float16* dp = &As[r * AS_STRIDE + c0];
            float vals[8] = {va.x, va.y, va.z, va.w, vb.x, vb.y, vb.z, vb.w};
#pragma unroll
            for (int j = 0; j < 8; j++) dp[j] = (_Float16)(vals[j] * s);
        }

#if HAVE_TDM
        // --- async: prefetch next weight tile, then wait for current one.
        if (wave == 0) {
            if (kb + 32 < Ktot) {
                tdm_load_b_tile(W + kb + 32, &Bs[buf ^ 1][0], Ktot);
                __builtin_amdgcn_s_wait_tensorcnt(1);  // TDM in-order: current done
            } else {
                __builtin_amdgcn_s_wait_tensorcnt(0);
            }
        }
#else
        // --- fallback: each thread copies one n-row (32 halves) of the tile
        {
            const _Float16* wp = W + (long)tid * Ktot + kb;
            v8h w0 = *(const v8h*)(wp + 0);
            v8h w1 = *(const v8h*)(wp + 8);
            v8h w2 = *(const v8h*)(wp + 16);
            v8h w3 = *(const v8h*)(wp + 24);
            _Float16* bp = &Bs[buf][tid * BS_STRIDE];
            *(v8h*)(bp + 0)  = w0;
            *(v8h*)(bp + 8)  = w1;
            *(v8h*)(bp + 16) = w2;
            *(v8h*)(bp + 24) = w3;
        }
#endif
        __syncthreads();

        // --- A fragment (16-bit A 16x32 layout: lane m, half-groups by hi)
        const _Float16* arow = &As[(mt * 16 + m) * AS_STRIDE];
        v8h alo = *(const v8h*)(arow + 8 * hi);
        v8h ahi = *(const v8h*)(arow + 16 + 8 * hi);
        v16h afrag;
#pragma unroll
        for (int i = 0; i < 8; i++) { afrag[i] = alo[i]; afrag[i + 8] = ahi[i]; }

        // --- pre-load ALL 8 B fragments into distinct regs so the DS loads
        //     pipeline instead of serializing one s_wait_dscnt per WMMA.
        v16h bfrag[8];
#pragma unroll
        for (int nt = 0; nt < 8; nt++) {
            const int col = nh * 128 + nt * 16 + m;
            const _Float16* brow = &Bs[buf][col * BS_STRIDE + 8 * hi];
            v8h blo = *(const v8h*)brow;
            v8h bhi = *(const v8h*)(brow + 16);
#pragma unroll
            for (int i = 0; i < 8; i++) { bfrag[nt][i] = blo[i]; bfrag[nt][i + 8] = bhi[i]; }
        }
        // --- back-to-back WMMA chain (independent accumulators)
#pragma unroll
        for (int nt = 0; nt < 8; nt++) {
            acc[nt] = __builtin_amdgcn_wmma_f32_16x16x32_f16(
                false, afrag, false, bfrag[nt], (short)0, acc[nt], false, false);
        }
        __syncthreads();
    }

    // --- store (C layout: vgpr r -> M = r + 8*hi, N = lane&15) + bias
#pragma unroll
    for (int nt = 0; nt < 8; nt++) {
        const int col = nh * 128 + nt * 16 + m;
        const float b = bias[col];
#pragma unroll
        for (int rr = 0; rr < 8; rr++) {
            int row = blockRow + mt * 16 + rr + 8 * hi;
            if (row < nrows) out[(long)row * HIDN + col] = acc[nt][rr] + b;
        }
    }
}

// ---------------- row epilogue: L2-normalize (+ BN eval + ReLU) in place ----
__global__ void row_epilogue(float* __restrict__ h,
                             const float* __restrict__ gamma, const float* __restrict__ beta,
                             const float* __restrict__ mean, const float* __restrict__ var,
                             int do_bn) {
    __shared__ float red[256];
    const int row = blockIdx.x;
    const int t   = threadIdx.x;
    float v = h[(long)row * HIDN + t];
    red[t] = v * v;
    __syncthreads();
#pragma unroll
    for (int s = 128; s > 0; s >>= 1) {
        if (t < s) red[t] += red[t + s];
        __syncthreads();
    }
    float inv = 1.0f / fmaxf(sqrtf(red[0]), 1e-12f);
    v *= inv;
    if (do_bn) {
        float g = gamma[t] * rsqrtf(var[t] + 1e-5f);
        v = (v - mean[t]) * g + beta[t];
        v = fmaxf(v, 0.0f);   // ReLU
    }
    h[(long)row * HIDN + t] = v;
}

// ---------------------------------------------------------------------------
extern "C" void kernel_launch(void* const* d_in, const int* in_sizes, int n_in,
                              void* d_out, int out_size, void* d_ws, size_t ws_size,
                              hipStream_t stream) {
    const float*     x   = (const float*)d_in[0];
    const long long* ei  = (const long long*)d_in[1];
    const float*     Wl0 = (const float*)d_in[2];
    const float*     bl0 = (const float*)d_in[3];
    const float*     Wr0 = (const float*)d_in[4];
    const float*     Wl1 = (const float*)d_in[5];
    const float*     bl1 = (const float*)d_in[6];
    const float*     Wr1 = (const float*)d_in[7];
    const float*     gmm = (const float*)d_in[8];
    const float*     bta = (const float*)d_in[9];
    const float*     mu  = (const float*)d_in[10];
    const float*     var = (const float*)d_in[11];
    float* out = (float*)d_out;

    const int IN_DIM = 128;
    const int N = in_sizes[0] / IN_DIM;  // 50000
    const int E = in_sizes[1] / 2;       // 640000

    // workspace carve-out
    char* base = (char*)d_ws;
    size_t off = 0;
    auto take = [&](size_t bytes) -> char* {
        char* p = base + off;
        off = (off + bytes + 255) & ~(size_t)255;
        return p;
    };
    _Float16* w0  = (_Float16*)take((size_t)256 * 256 * 2);   // Wt0 [256][256]
    _Float16* w1  = (_Float16*)take((size_t)256 * 512 * 2);   // Wt1 [256][512]
    float*    cnt = (float*)take((size_t)N * 4);
    float*    rs  = (float*)take((size_t)N * 4);
    float*    agg = (float*)take((size_t)N * 256 * 4);
    float*    h   = (float*)take((size_t)N * 256 * 4);

    const int T = 256;
    // graph stats (shared by both layers)
    zero_f32<<<(N + T - 1) / T, T, 0, stream>>>(cnt, N);
    zero_f32<<<(int)(((long)N * 128 + T - 1) / T), T, 0, stream>>>(agg, (long)N * 128);
    cvt_weights_t<<<(256 * 256 + T - 1) / T, T, 0, stream>>>(Wl0, Wr0, w0, 128, 128);
    cvt_weights_t<<<(512 * 256 + T - 1) / T, T, 0, stream>>>(Wl1, Wr1, w1, 256, 256);
    edge_count<<<(E + T - 1) / T, T, 0, stream>>>(ei, cnt, E);
    make_rscale<<<(N + T - 1) / T, T, 0, stream>>>(cnt, rs, N);

    // ---- layer 0 ----
    scatter_add<128><<<(int)(((long)E * 32 + T - 1) / T), T, 0, stream>>>(ei, x, agg, E);
    sage_gemm<<<(N + 63) / 64, T, 0, stream>>>(agg, 128, x, 128, rs, w0, bl0, h, 128, 128, N);
    row_epilogue<<<N, T, 0, stream>>>(h, gmm, bta, mu, var, 1);

    // ---- layer 1 ----
    zero_f32<<<(int)(((long)N * 256 + T - 1) / T), T, 0, stream>>>(agg, (long)N * 256);
    scatter_add<256><<<(int)(((long)E * 64 + T - 1) / T), T, 0, stream>>>(ei, h, agg, E);
    sage_gemm<<<(N + 63) / 64, T, 0, stream>>>(agg, 256, h, 256, rs, w1, bl1, out, 256, 256, N);
    row_epilogue<<<N, T, 0, stream>>>(out, nullptr, nullptr, nullptr, nullptr, 0);
}
